// AttentionLayer_7335804142263
// MI455X (gfx1250) — matrix operand order repaired and checked by reading, not compiled, not run on hardware
//
#include <hip/hip_runtime.h>
#include <stdint.h>

#define BB 32
#define TT 2048
#define DD 1024
#define AA 1024

typedef __bf16    v16bf __attribute__((ext_vector_type(16)));
typedef float     v8f   __attribute__((ext_vector_type(8)));
typedef uint32_t  v8u   __attribute__((ext_vector_type(8)));

static __device__ __forceinline__ uint32_t f32_to_bf16_rne(float f) {
  uint32_t u = __builtin_bit_cast(uint32_t, f);
  uint32_t r = u + 0x7FFFu + ((u >> 16) & 1u);
  return r >> 16;
}
static __device__ __forceinline__ uint32_t pack_bf16(float lo, float hi) {
  return f32_to_bf16_rne(lo) | (f32_to_bf16_rne(hi) << 16);
}

// ---------------------------------------------------------------------------
// Kernel 1: convert U_a [D, A] f32 -> bf16 laid out in WMMA B-fragment order.
// Layout: ub[((nt*32 + kt)*32 + lane)*8 + j]
//   B (32x16 bf16) fragment layout (contiguous k order, as for A):
//     n = nt*16 + (lane & 15)
//     k = kt*32 + (lane>=16 ? 16 : 0) + 2*j   (pair packed lo/hi in dword)
// ---------------------------------------------------------------------------
__global__ void uconv_kernel(const float* __restrict__ U, uint32_t* __restrict__ ub) {
  int tid  = blockIdx.x * blockDim.x + threadIdx.x;   // 0 .. 524287
  int j    = tid & 7;
  int lane = (tid >> 3) & 31;
  int frag = tid >> 8;            // nt*32 + kt
  int kt   = frag & 31;
  int nt   = frag >> 5;
  int k    = kt * 32 + ((lane >= 16) ? 16 : 0) + 2 * j;
  int n    = nt * 16 + (lane & 15);
  float lo = U[(size_t)k * AA + n];
  float hi = U[(size_t)(k + 1) * AA + n];
  ub[tid]  = pack_bf16(lo, hi);
}

// ---------------------------------------------------------------------------
// Kernel 2: W_a_s[b, a] = sum_k s[b,k] * W_a[k,a]   (tiny, f32)
// ---------------------------------------------------------------------------
__global__ void was_kernel(const float* __restrict__ s, const float* __restrict__ W,
                           float* __restrict__ was) {
  int a = blockIdx.x * blockDim.x + threadIdx.x;  // 0..1023
  int b = blockIdx.y;
  float acc = 0.f;
  for (int k = 0; k < DD; ++k)
    acc += s[b * DD + k] * W[(size_t)k * AA + a];
  was[b * AA + a] = acc;
}

// ---------------------------------------------------------------------------
// Async copy of one 32KB B-slab (8192 dwords) into LDS, 128 threads.
// Each thread issues 16 x global_load_async_to_lds_b128 (ASYNCcnt-tracked).
// INST_OFFSET is added to BOTH the LDS and global addresses (ISA 08 §4.4),
// so consecutive issues stride 2048B (= 128 lanes * 16B) on both sides.
// ---------------------------------------------------------------------------
#define ASYNC_CP(OFF)                                                      \
  asm volatile("global_load_async_to_lds_b128 %0, %1, %2 offset:" #OFF     \
               :: "v"(lds), "v"(voff), "s"(gbase) : "memory");

static __device__ __forceinline__ void issue_bslab_async(
    const uint32_t* __restrict__ ub, int nt, uint32_t lds_base, int tid) {
  uint64_t gbase = (uint64_t)(uintptr_t)(ub + (size_t)nt * 8192);
  uint32_t voff  = (uint32_t)(tid * 16);
  uint32_t lds   = lds_base + (uint32_t)(tid * 16);
  ASYNC_CP(0)     ASYNC_CP(2048)  ASYNC_CP(4096)  ASYNC_CP(6144)
  ASYNC_CP(8192)  ASYNC_CP(10240) ASYNC_CP(12288) ASYNC_CP(14336)
  ASYNC_CP(16384) ASYNC_CP(18432) ASYNC_CP(20480) ASYNC_CP(22528)
  ASYNC_CP(24576) ASYNC_CP(26624) ASYNC_CP(28672) ASYNC_CP(30720)
}

// ---------------------------------------------------------------------------
// Kernel 3: fused  scores[b,t] = sum_a tanh( (h@U_a)[b,t,a] + was[b,a] ) * v_a[a]
// Block: 128 threads (4 waves). Block owns 64 consecutive T rows of one b.
// Wave w owns the 16-row M-tile [w*16, w*16+16).
// Key idea: each wave's 16 h-rows are invariant over the 64 N-tiles, so ALL
// 32 A-fragments (256 VGPRs) are loaded from LDS ONCE and kept in registers;
// the hot loop reads only the B fragment (32B/lane per wmma), matching LDS
// bank bandwidth. B slabs are double-buffered via global_load_async_to_lds.
// ---------------------------------------------------------------------------
#define HPITCH 516   // dwords per LDS row: 512 data + 4 pad (kills bank conflicts)

__global__ __launch_bounds__(128) void scores_kernel(
    const float* __restrict__ h, const uint32_t* __restrict__ ub,
    const float* __restrict__ was, const float* __restrict__ va,
    float* __restrict__ scores) {
  __shared__ uint32_t hbf[64 * HPITCH];        // 132,096 B
  __shared__ uint32_t bslab[2][32 * 32 * 8];   // 2 x 32,768 B

  const int tid  = threadIdx.x;
  const int lane = tid & 31;
  const int wave = tid >> 5;          // 0..3
  const int blk  = blockIdx.x;        // 0..1023
  const int b    = blk >> 4;
  const int t0   = (blk & 15) * 64;

  // LDS byte offsets of the two B buffers (low 32 bits of generic LDS ptr)
  const uint32_t bs0 = (uint32_t)(uintptr_t)&bslab[0][0];
  const uint32_t bs1 = (uint32_t)(uintptr_t)&bslab[1][0];

  // Kick off the nt=0 slab copy; it overlaps the whole h-staging phase.
  issue_bslab_async(ub, 0, bs0, tid);

  // ---- stage h[b, t0:t0+64, :] f32 -> bf16 pairs in LDS (coalesced) ----
  const float* hb = h + ((size_t)b * TT + t0) * DD;
  for (int idx = tid; idx < 64 * 512; idx += 128) {    // exact multiple, no divergence
    int row = idx >> 9;
    int cp  = idx & 511;               // column pair
    float2 v = *(const float2*)(hb + (size_t)row * DD + cp * 2);
    hbf[row * HPITCH + cp] = pack_bf16(v.x, v.y);
  }
  __syncthreads();

  const int mrow0 = wave * 16;
  const int rlo   = lane & 15;
  const int hi    = lane >> 4;         // 0 or 1

  // ---- hoist ALL A fragments (32 K-tiles x 8 dwords) into registers ----
  // A (16x32 bf16) fragment, contiguous k order (same as B):
  //   lanes 0-15: row = lane,    K = {0..15}   (+kt*32)
  //   lanes16-31: row = lane-16, K = {16..31}
  union AF { v8u u; v16bf bf; };
  AF a[32];
  {
    const uint32_t* arow = &hbf[(mrow0 + rlo) * HPITCH];
#pragma unroll
    for (int kt = 0; kt < 32; ++kt) {
      int kb = kt * 16 + (hi ? 8 : 0);
      uint4 a0 = *(const uint4*)(arow + kb);
      uint4 a1 = *(const uint4*)(arow + kb + 4);
      a[kt].u[0] = a0.x; a[kt].u[1] = a0.y; a[kt].u[2] = a0.z; a[kt].u[3] = a0.w;
      a[kt].u[4] = a1.x; a[kt].u[5] = a1.y; a[kt].u[6] = a1.z; a[kt].u[7] = a1.w;
    }
  }

  float sacc[8];
#pragma unroll
  for (int r = 0; r < 8; ++r) sacc[r] = 0.f;

  for (int nt = 0; nt < 64; ++nt) {
    const int cur = nt & 1;
    // Prefetch the next slab into the spare buffer (freed by the barrier at
    // the end of the previous iteration), then wait only for the CURRENT
    // slab: async loads complete in order, so <=16 outstanding means the
    // previous 16-op batch (this nt's slab) has fully landed in LDS.
    if (nt < 63) {
      issue_bslab_async(ub, nt + 1, cur ? bs0 : bs1, tid);
      asm volatile("s_wait_asynccnt 0x10" ::: "memory");
    } else {
      asm volatile("s_wait_asynccnt 0x0" ::: "memory");
    }
    __syncthreads();                       // slab[cur] visible to all waves

    v8f acc = {};
    const uint32_t* bbase = &bslab[cur][0];
#pragma unroll
    for (int kt = 0; kt < 32; ++kt) {
      union { v8u u; v16bf bf; } bfrag;
      const uint32_t* bsrc = bbase + (kt * 32 + lane) * 8;
      uint4 b0 = *(const uint4*)bsrc;
      uint4 b1 = *(const uint4*)(bsrc + 4);
      bfrag.u[0] = b0.x; bfrag.u[1] = b0.y; bfrag.u[2] = b0.z; bfrag.u[3] = b0.w;
      bfrag.u[4] = b1.x; bfrag.u[5] = b1.y; bfrag.u[6] = b1.z; bfrag.u[7] = b1.w;
      acc = __builtin_amdgcn_wmma_f32_16x16x32_bf16(
          /*neg_a=*/false, a[kt].bf, /*neg_b=*/false, bfrag.bf,
          /*c_mod=*/(short)0, acc, /*reuse_a=*/false, /*reuse_b=*/false);
    }

    // C/D layout: VGPR r, lanes0-15 -> (M=r, N=lane), lanes16-31 -> (M=r+8, N=lane-16)
    int nn   = nt * 16 + rlo;
    float wv = was[b * AA + nn];
    float vv = va[nn];
#pragma unroll
    for (int r = 0; r < 8; ++r) {
      float e = tanhf(acc[r] + wv);
      sacc[r] += e * vv;
    }
    __syncthreads();                       // all waves done reading slab[cur]
  }

  // reduce over N: sum across each 16-lane half
#pragma unroll
  for (int r = 0; r < 8; ++r) {
    float v = sacc[r];
    v += __shfl_xor(v, 8, 16);
    v += __shfl_xor(v, 4, 16);
    v += __shfl_xor(v, 2, 16);
    v += __shfl_xor(v, 1, 16);
    sacc[r] = v;
  }
  if (rlo == 0) {
    int m = mrow0 + hi * 8;
#pragma unroll
    for (int r = 0; r < 8; ++r)
      scores[(size_t)b * TT + t0 + m + r] = sacc[r];
  }
}

// ---------------------------------------------------------------------------
// Kernel 4: raw exp (reference does NOT max-subtract) + per-b sum
// ---------------------------------------------------------------------------
__global__ __launch_bounds__(256) void softmax_kernel(
    const float* __restrict__ scores, float* __restrict__ expsc,
    float* __restrict__ denom) {
  __shared__ float red[256];
  int b = blockIdx.x, tid = threadIdx.x;
  float acc = 0.f;
  for (int t = tid; t < TT; t += 256) {
    float e = expf(scores[b * TT + t]);
    expsc[b * TT + t] = e;
    acc += e;
  }
  red[tid] = acc;
  __syncthreads();
  for (int s = 128; s > 0; s >>= 1) {
    if (tid < s) red[tid] += red[tid + s];
    __syncthreads();
  }
  if (tid == 0) denom[b] = red[0];
}

// ---------------------------------------------------------------------------
// Kernel 5: c[b,d] = (1/denom) * sum_t expsc[b,t] * h[b,t,d]  (coalesced over d)
// ---------------------------------------------------------------------------
__global__ __launch_bounds__(256) void context_kernel(
    const float* __restrict__ h, const float* __restrict__ expsc,
    const float* __restrict__ denom, float* __restrict__ out) {
  __shared__ float w[TT];
  int b = blockIdx.y;
  int d = blockIdx.x * 256 + threadIdx.x;
  for (int t = threadIdx.x; t < TT; t += 256)
    w[t] = expsc[b * TT + t];
  __syncthreads();
  float inv = 1.0f / denom[b];
  const float* hb = h + (size_t)b * TT * DD + d;
  float a0 = 0.f, a1 = 0.f, a2 = 0.f, a3 = 0.f;
  for (int t = 0; t < TT; t += 4) {
    a0 += w[t + 0] * hb[(size_t)(t + 0) * DD];
    a1 += w[t + 1] * hb[(size_t)(t + 1) * DD];
    a2 += w[t + 2] * hb[(size_t)(t + 2) * DD];
    a3 += w[t + 3] * hb[(size_t)(t + 3) * DD];
  }
  out[b * DD + d] = (a0 + a1 + a2 + a3) * inv;
}

// ---------------------------------------------------------------------------
extern "C" void kernel_launch(void* const* d_in, const int* in_sizes, int n_in,
                              void* d_out, int out_size, void* d_ws, size_t ws_size,
                              hipStream_t stream) {
  const float* s   = (const float*)d_in[0];   // [B, D]
  const float* h   = (const float*)d_in[1];   // [B, T, D]
  const float* W_a = (const float*)d_in[2];   // [D, A]
  const float* U_a = (const float*)d_in[3];   // [D, A]
  const float* v_a = (const float*)d_in[4];   // [A]
  float* out = (float*)d_out;                 // [B, D]

  char* ws = (char*)d_ws;
  uint32_t* ub  = (uint32_t*)(ws);                                      // 2 MB
  float*  was   = (float*)(ws + (2u << 20));                            // 128 KB
  float*  scores= (float*)(ws + (2u << 20) + (128u << 10));             // 256 KB
  float*  expsc = (float*)(ws + (2u << 20) + (384u << 10));             // 256 KB
  float*  denom = (float*)(ws + (2u << 20) + (640u << 10));             // 128 B

  uconv_kernel  <<<2048, 256, 0, stream>>>(U_a, ub);
  was_kernel    <<<dim3(AA / 256, BB), 256, 0, stream>>>(s, W_a, was);
  scores_kernel <<<BB * (TT / 64), 128, 0, stream>>>(h, ub, was, v_a, scores);
  softmax_kernel<<<BB, 256, 0, stream>>>(scores, expsc, denom);
  context_kernel<<<dim3(DD / 256, BB), 256, 0, stream>>>(h, expsc, denom, out);
}
